// Int8Linear_5222680232429
// MI455X (gfx1250) — 8-bit matrix operand order repaired and checked by reading, compiled, not run on hardware
//
#include <hip/hip_runtime.h>

typedef int v8i __attribute__((ext_vector_type(8)));
typedef int v4i __attribute__((ext_vector_type(4)));


__global__ void __launch_bounds__(256)
quantize_pack_kernel(const float* __restrict__ in, signed char* __restrict__ qp,
                     float* __restrict__ scale, int K, int is_b)
{
    const int row = blockIdx.x;
    const int t   = threadIdx.x;
    const float4* rp = (const float4*)(in + (size_t)row * K);
    const int n4 = K >> 2;

    const bool cached = (n4 == 1024);
    float4 vals[4];

    float amax = 0.0f;
    if (cached) {
        #pragma unroll
        for (int j = 0; j < 4; ++j) {
            float4 v = rp[t + j * 256];
            vals[j] = v;
            amax = fmaxf(amax, fmaxf(fmaxf(fabsf(v.x), fabsf(v.y)),
                                     fmaxf(fabsf(v.z), fabsf(v.w))));
        }
    } else {
        for (int i = t; i < n4; i += 256) {
            float4 v = rp[i];
            amax = fmaxf(amax, fmaxf(fmaxf(fabsf(v.x), fabsf(v.y)),
                                     fmaxf(fabsf(v.z), fabsf(v.w))));
        }
    }

    __shared__ float red[256];
    red[t] = amax;
    __syncthreads();
    #pragma unroll
    for (int s = 128; s > 0; s >>= 1) {
        if (t < s) red[t] = fmaxf(red[t], red[t + s]);
        __syncthreads();
    }
    const float a   = fmaxf(red[0], 1e-8f);
    const float inv = 127.0f / a;
    if (t == 0) scale[row] = a * (1.0f / 127.0f);

    const int r16      = row & 15;
    const size_t tilerow = (size_t)(row >> 4) * (size_t)(K >> 6);
    #pragma unroll 4
    for (int i = t; i < n4; i += 256) {
        float4 v;
        if (cached) v = vals[(i - t) >> 8];
        else        v = rp[i];

        int ix = __float2int_rn(v.x * inv);
        int iy = __float2int_rn(v.y * inv);
        int iz = __float2int_rn(v.z * inv);
        int iw = __float2int_rn(v.w * inv);
        ix = min(127, max(-127, ix));
        iy = min(127, max(-127, iy));
        iz = min(127, max(-127, iz));
        iw = min(127, max(-127, iw));
        char4 c;
        c.x = (signed char)ix; c.y = (signed char)iy;
        c.z = (signed char)iz; c.w = (signed char)iw;

        const int k  = i << 2;
        const int kb = k >> 6;
        const int kk = k & 63;
        int lane, within;
        if (is_b) {
            lane   = r16 + (((kk >> 4) & 1) << 4);
            within = kk & 15;
        } else {
            lane   = r16 + (((kk >> 4) & 1) << 4);
            within = kk & 15;
        }
        const int chunk = kk >> 5;
        const size_t off = (tilerow + kb) * 1024 + chunk * 512 + lane * 16 + within;
        *(char4*)(qp + off) = c;
    }
}

__global__ void __launch_bounds__(256)
gemm_i8_wmma_kernel(const signed char* __restrict__ xq,
                    const signed char* __restrict__ wq,
                    const float* __restrict__ sx,
                    const float* __restrict__ sw,
                    const float* __restrict__ bias,
                    float* __restrict__ out,
                    int M, int N, int K)
{
    const int lane = threadIdx.x & 31;
    const int wave = threadIdx.x >> 5;
    const int wm   = wave & 1;
    const int wn   = wave >> 1;
    const int mt0  = blockIdx.x * 8 + wm * 4;
    const int nt0  = blockIdx.y * 16 + wn * 4;
    const int ktiles = K >> 6;
    const size_t tstride = (size_t)ktiles * 1024;

    const signed char* abase = xq + (size_t)mt0 * tstride + lane * 16;
    const signed char* bbase = wq + (size_t)nt0 * tstride + lane * 16;

    v8i acc[4][4];
    #pragma unroll
    for (int mt = 0; mt < 4; ++mt)
        #pragma unroll
        for (int nt = 0; nt < 4; ++nt)
            acc[mt][nt] = (v8i)(0);

    for (int kb = 0; kb < ktiles; ++kb) {
        const size_t koff = (size_t)kb * 1024;
        v8i afrag[4], bfrag[4];
        #pragma unroll
        for (int mt = 0; mt < 4; ++mt) {
            const signed char* p = abase + mt * tstride + koff;
            v4i f0 = *(const v4i*)(p);
            v4i f1 = *(const v4i*)(p + 512);
            v8i f;
            f[0] = f0.x; f[1] = f0.y; f[2] = f0.z; f[3] = f0.w;
            f[4] = f1.x; f[5] = f1.y; f[6] = f1.z; f[7] = f1.w;
            afrag[mt] = f;
        }
        #pragma unroll
        for (int nt = 0; nt < 4; ++nt) {
            const signed char* p = bbase + nt * tstride + koff;
            v4i f0 = *(const v4i*)(p);
            v4i f1 = *(const v4i*)(p + 512);
            v8i f;
            f[0] = f0.x; f[1] = f0.y; f[2] = f0.z; f[3] = f0.w;
            f[4] = f1.x; f[5] = f1.y; f[6] = f1.z; f[7] = f1.w;
            bfrag[nt] = f;
        }
        #pragma unroll
        for (int mt = 0; mt < 4; ++mt)
            #pragma unroll
            for (int nt = 0; nt < 4; ++nt)
                acc[mt][nt] = __builtin_amdgcn_wmma_i32_16x16x64_iu8(
                     true, afrag[mt],
                     true, bfrag[nt],
                    acc[mt][nt],  false,  false);
    }

    const int half = lane >> 4;
    const int l16  = lane & 15;
    const int wave_m = mt0 << 4;
    const int wave_n = nt0 << 4;

    float sxv[4][8];
    #pragma unroll
    for (int mt = 0; mt < 4; ++mt)
        #pragma unroll
        for (int v = 0; v < 8; ++v)
            sxv[mt][v] = sx[wave_m + mt * 16 + half * 8 + v];

    #pragma unroll
    for (int nt = 0; nt < 4; ++nt) {
        const int col = wave_n + nt * 16 + l16;
        const float swv = sw[col];
        const float bv  = bias[col];
        #pragma unroll
        for (int mt = 0; mt < 4; ++mt) {
            const int rbase = wave_m + mt * 16 + half * 8;
            #pragma unroll
            for (int v = 0; v < 8; ++v) {
                out[(size_t)(rbase + v) * N + col] =
                    (float)acc[mt][nt][v] * sxv[mt][v] * swv + bv;
            }
        }
    }
}

extern "C" void kernel_launch(void* const* d_in, const int* in_sizes, int n_in,
                              void* d_out, int out_size, void* d_ws, size_t ws_size,
                              hipStream_t stream)
{
    const float* x    = (const float*)d_in[0];
    const float* w    = (const float*)d_in[1];
    const float* bias = (const float*)d_in[2];
    float* out = (float*)d_out;

    const int N = in_sizes[2];
    const int K = in_sizes[1] / N;
    const int M = in_sizes[0] / K;

    signed char* xq = (signed char*)d_ws;
    signed char* wq = xq + (size_t)M * K;
    float* sx = (float*)(wq + (size_t)N * K);
    float* sw = sx + M;

    quantize_pack_kernel<<<M, 256, 0, stream>>>(x, xq, sx, K,  0);
    quantize_pack_kernel<<<N, 256, 0, stream>>>(w, wq, sw, K,  1);

    dim3 grid((M + 127) / 128, (N + 255) / 256);
    gemm_i8_wmma_kernel<<<grid, 256, 0, stream>>>(xq, wq, sx, sw, bias, out,
                                                  M, N, K);
}
